// Model_56212531970604
// MI455X (gfx1250) — matrix operand order repaired and checked by reading, not compiled, not run on hardware
//
#include <hip/hip_runtime.h>
#include <hip/hip_bf16.h>

// ---------------------------------------------------------------------------
// Pokemon battle-value network forward pass for gfx1250 (MI455X).
// All large GEMMs run on v_wmma_f32_16x16x32_f16 (f16 operands, f32 accum).
// Weights are re-packed (f32 -> f16, WMMA-fragment-linear) per launch.
// GEMM-feeding activations are stored f16 (halves HBM/L2 traffic at 23.3TB/s
// and removes f32->f16 cvt from the WMMA inner loop); the transformer residual
// stream stays f32. Each wave register-blocks 4 N-tiles (NT=4) to reuse the
// A fragment across 4 WMMAs and cut redundant A traffic 4x.
// ---------------------------------------------------------------------------

typedef __attribute__((ext_vector_type(16))) _Float16 v16h;
typedef __attribute__((ext_vector_type(8)))  _Float16 v8h;
typedef __attribute__((ext_vector_type(8)))  float    v8f;

#define BB   8192          // batch
#define SS   12            // tokens per sample
#define PP   (BB * SS)     // 98304 rows through transformer
#define RR   (BB * 48)     // 393216 move rows
#define EPSF 1e-5f

__device__ __forceinline__ float wave_sum(float v) {
#pragma unroll
  for (int off = 16; off > 0; off >>= 1) v += __shfl_xor(v, off, 32);
  return v;
}

// ---------------------------------------------------------------------------
// Weight packing: f32 [K,N] row-major  ->  f16 fragment-linear layout.
// Fragment (kTile,nTile) holds a 32x16 B-operand; per lane l:
//   n = nTile*16 + (l&15), k = kTile*32 + (l>=16 ? 8 : 0) + (i<8 ? i : i+8), i = 0..15
// stored as 16 contiguous halves at F[((kTile*nTiles+nTile)*32 + l)*16].
// Rows k >= Kreal are zero-padded (poke_W 567 -> 576).
// ---------------------------------------------------------------------------
__global__ void pack_weight_kernel(const float* __restrict__ W,
                                   _Float16* __restrict__ F,
                                   int K, int N, int Kreal) {
  int tid = blockIdx.x * 256 + threadIdx.x;
  int nTiles = N >> 4;
  int total = (K >> 5) * nTiles * 32;
  if (tid >= total) return;
  int lane = tid & 31;
  int frag = tid >> 5;
  int kTile = frag / nTiles, nTile = frag % nTiles;
  int n = (nTile << 4) + (lane & 15);
  int kbase = (kTile << 5) + ((lane >> 4) << 3);
  _Float16 vals[16];
#pragma unroll
  for (int i = 0; i < 16; i++) {
    int k = kbase + (i < 8 ? i : i + 8);
    float w = (k < Kreal) ? W[(size_t)k * N + n] : 0.0f;
    vals[i] = (_Float16)w;
  }
  *(v16h*)(F + (size_t)tid * 16) = *(v16h*)vals;
}

// Pack per-head QKV weights [H=8, D=512, HD=64] as logical [K=512, N=512],
// N = h*64 + c  (matches o.transpose(0,2,1,3).reshape concat order).
__global__ void pack_qkv_kernel(const float* __restrict__ Wh,
                                _Float16* __restrict__ F) {
  int tid = blockIdx.x * 256 + threadIdx.x;
  const int nTiles = 32;
  int total = 16 * nTiles * 32;
  if (tid >= total) return;
  int lane = tid & 31;
  int frag = tid >> 5;
  int kTile = frag / nTiles, nTile = frag % nTiles;
  int n = (nTile << 4) + (lane & 15);
  int h = n >> 6, c = n & 63;
  int kbase = (kTile << 5) + ((lane >> 4) << 3);
  _Float16 vals[16];
#pragma unroll
  for (int i = 0; i < 16; i++) {
    int k = kbase + (i < 8 ? i : i + 8);
    vals[i] = (_Float16)Wh[((size_t)(h * 512 + k)) * 64 + c];
  }
  *(v16h*)(F + (size_t)tid * 16) = *(v16h*)vals;
}

// ---------------------------------------------------------------------------
// WMMA GEMM:  out[M,N] = epi( A[M,K](f16) * Wfrag(f16) + bias [+resid(f32)] )
// One wave per 16x(16*NT) output strip, 8 waves per block, K-step 32.
// A-fragment (ISA 16-bit layout): lane l holds row M=l&15, K offsets
//   {koff..koff+7, koff+16..koff+23}, koff=(l>=16)*8 -> two 16B loads.
// Epilogue writes f32 (C32) and/or f16 (C16).
// M%16==0, K%32==0, N%(16*NT)==0 at every call site.
// ---------------------------------------------------------------------------
template <int NT>
__global__ void gemm_f16_kernel(const _Float16* __restrict__ A,
                                const _Float16* __restrict__ F,
                                const float* __restrict__ bias,
                                const float* __restrict__ resid,
                                float* __restrict__ C32,
                                _Float16* __restrict__ C16,
                                int M, int N, int K, int doRelu) {
  int lane = threadIdx.x & 31;
  int wave = threadIdx.x >> 5;
  int nTiles = N >> 4;
  int nGroups = nTiles / NT;
  long tile = (long)blockIdx.x * 8 + wave;
  long mTile = tile / nGroups;
  int g = (int)(tile % nGroups);
  if (mTile >= (long)(M >> 4)) return;   // wave-uniform: EXEC all-ones at WMMA
  int nTile0 = g * NT;

  int mRow = ((int)mTile << 4) + (lane & 15);
  int koff = (lane >> 4) << 3;           // 0 or 8
  const _Float16* arow = A + (size_t)mRow * K;
  const _Float16* fb = F + ((size_t)nTile0 * 32 + lane) * 16;
  size_t fstep = (size_t)nTiles * 512;   // halves per kTile

  v8f acc[NT];
#pragma unroll
  for (int t = 0; t < NT; t++) acc[t] = (v8f){0.f,0.f,0.f,0.f,0.f,0.f,0.f,0.f};

  for (int k0 = 0; k0 < K; k0 += 32) {
    v8h alo = *(const v8h*)(arow + k0 + koff);        // K = koff .. koff+7
    v8h ahi = *(const v8h*)(arow + k0 + koff + 16);   // K = koff+16 .. koff+23
    v16h af = __builtin_shufflevector(alo, ahi,
        0, 1, 2, 3, 4, 5, 6, 7, 8, 9, 10, 11, 12, 13, 14, 15);
#pragma unroll
    for (int t = 0; t < NT; t++) {
      v16h bf = *(const v16h*)(fb + (size_t)t * 512);
      acc[t] = __builtin_amdgcn_wmma_f32_16x16x32_f16(
          false, af, false, bf, (short)0, acc[t], false, false);
    }
    fb += fstep;
  }

  // C/D layout: lane<16 -> col=lane, rows 0..7; lane>=16 -> col=lane-16, rows 8..15
  int rbase = ((int)mTile << 4) + ((lane >> 4) << 3);
#pragma unroll
  for (int t = 0; t < NT; t++) {
    int col = ((nTile0 + t) << 4) + (lane & 15);
    float bv = bias ? bias[col] : 0.0f;
#pragma unroll
    for (int v = 0; v < 8; v++) {
      size_t idx = (size_t)(rbase + v) * N + col;
      float val = acc[t][v] + bv;
      if (resid) val += resid[idx];
      if (doRelu) val = fmaxf(val, 0.0f);
      if (C32) C32[idx] = val;
      if (C16) C16[idx] = (_Float16)val;
    }
  }
}

// ---------------------------------------------------------------------------
// LayerNorm over f32 rows of length Dn, f16 output (one wave32 per row).
// ---------------------------------------------------------------------------
__global__ void layernorm_kernel(const float* __restrict__ X,
                                 const float* __restrict__ g,
                                 const float* __restrict__ b,
                                 _Float16* __restrict__ Y, int Dn) {
  int row = blockIdx.x * 8 + (threadIdx.x >> 5);
  int lane = threadIdx.x & 31;
  const float* x = X + (size_t)row * Dn;
  float s = 0.f;
  for (int i = lane; i < Dn; i += 32) s += x[i];
  float mu = wave_sum(s) / (float)Dn;
  float v = 0.f;
  for (int i = lane; i < Dn; i += 32) { float d = x[i] - mu; v += d * d; }
  float rstd = rsqrtf(wave_sum(v) / (float)Dn + EPSF);
  _Float16* y = Y + (size_t)row * Dn;
  for (int i = lane; i < Dn; i += 32)
    y[i] = (_Float16)((x[i] - mu) * rstd * g[i] + b[i]);
}

// ---------------------------------------------------------------------------
// Attention per (sample, head): S=12, HD=64; tiny -> VALU + LDS, f16 I/O.
// ---------------------------------------------------------------------------
__global__ void attention_kernel(const _Float16* __restrict__ Q,
                                 const _Float16* __restrict__ Kx,
                                 const _Float16* __restrict__ V,
                                 _Float16* __restrict__ O) {
  int bh = blockIdx.x;             // b*8 + h
  int b = bh >> 3, h = bh & 7;
  __shared__ float qs[12][64], ks[12][64], vs[12][64], sc[12][12];
  size_t base = (size_t)b * 12 * 512 + (size_t)h * 64;
  int t = threadIdx.x;
  for (int i = t; i < 768; i += 256) {
    int s = i >> 6, d = i & 63;
    size_t idx = base + (size_t)s * 512 + d;
    qs[s][d] = (float)Q[idx]; ks[s][d] = (float)Kx[idx]; vs[s][d] = (float)V[idx];
  }
  __syncthreads();
  if (t < 144) {
    int qi = t / 12, ki = t % 12;
    float a = 0.f;
#pragma unroll
    for (int d = 0; d < 64; d++) a += qs[qi][d] * ks[ki][d];
    sc[qi][ki] = a * 0.125f;       // 1/sqrt(64)
  }
  __syncthreads();
  if (t < 12) {
    float m = -1e30f;
    for (int k = 0; k < 12; k++) m = fmaxf(m, sc[t][k]);
    float s = 0.f;
    for (int k = 0; k < 12; k++) { float e = __expf(sc[t][k] - m); sc[t][k] = e; s += e; }
    float inv = 1.f / s;
    for (int k = 0; k < 12; k++) sc[t][k] *= inv;
  }
  __syncthreads();
  for (int i = t; i < 768; i += 256) {
    int qi = i >> 6, d = i & 63;
    float a = 0.f;
#pragma unroll
    for (int k = 0; k < 12; k++) a += sc[qi][k] * vs[k][d];
    O[base + (size_t)qi * 512 + d] = (_Float16)a;
  }
}

// ---------------------------------------------------------------------------
// Board encoder: gather 35 feats -> LN -> 35x64 matmul -> relu -> xcat[:,1024:]
// ---------------------------------------------------------------------------
__global__ void board_kernel(const int* __restrict__ bi, const float* __restrict__ bf,
                             const float* __restrict__ twE, const float* __restrict__ trE,
                             const float* __restrict__ weE, const float* __restrict__ teE,
                             const float* __restrict__ lng, const float* __restrict__ lnb,
                             const float* __restrict__ W, const float* __restrict__ bb,
                             float* __restrict__ xcat) {
  int b = blockIdx.x;
  __shared__ float comb[35], nrm[35];
  int t = threadIdx.x;
  if (t < 35) {
    float v;
    if (t < 8)       v = twE[bi[b * 5 + (t >> 2)] * 4 + (t & 3)];
    else if (t < 12) v = trE[bi[b * 5 + 2] * 4 + (t - 8)];
    else if (t < 16) v = weE[bi[b * 5 + 3] * 4 + (t - 12)];
    else if (t < 20) v = teE[bi[b * 5 + 4] * 4 + (t - 16)];
    else             v = bf[b * 15 + (t - 20)];
    comb[t] = v;
  }
  __syncthreads();
  float mu = 0.f;
  for (int i = 0; i < 35; i++) mu += comb[i];
  mu *= (1.f / 35.f);
  float var = 0.f;
  for (int i = 0; i < 35; i++) { float d = comb[i] - mu; var += d * d; }
  float rstd = rsqrtf(var * (1.f / 35.f) + EPSF);
  if (t < 35) nrm[t] = (comb[t] - mu) * rstd * lng[t] + lnb[t];
  __syncthreads();
  float acc = bb[t];
  for (int k = 0; k < 35; k++) acc += nrm[k] * W[k * 64 + t];
  xcat[(size_t)b * 1088 + 1024 + t] = fmaxf(acc, 0.f);
}

// typeEmbU[20,64] = typeEmb[20,10] @ U[10,64]
__global__ void type_u_kernel(const float* __restrict__ tE, const float* __restrict__ U,
                              float* __restrict__ out) {
  int tid = blockIdx.x * 256 + threadIdx.x;
  if (tid >= 20 * 64) return;
  int t = tid >> 6, c = tid & 63;
  float a = 0.f;
#pragma unroll
  for (int j = 0; j < 10; j++) a += tE[t * 10 + j] * U[j * 64 + c];
  out[tid] = a;
}

// mcomb = relu(moveEmb[name] + typeEmbU[type] + moveFeats @ Wm) -> f16 [RR,64]
__global__ void mcomb_kernel(const int* __restrict__ mi, const float* __restrict__ mf,
                             const float* __restrict__ mE, const float* __restrict__ tU,
                             const float* __restrict__ Wm, _Float16* __restrict__ out) {
  long tid = (long)blockIdx.x * 256 + threadIdx.x;
  if (tid >= (long)RR * 64) return;
  long r = tid >> 6; int c = (int)(tid & 63);
  int mname = mi[r * 2 + 0], mtype = mi[r * 2 + 1];
  float acc = mE[mname * 64 + c] + tU[mtype * 64 + c];
  const float* f = mf + r * 6;
#pragma unroll
  for (int j = 0; j < 6; j++) acc += f[j] * Wm[j * 64 + c];
  out[tid] = (_Float16)fmaxf(acc, 0.f);
}

// moves[q,c] = sum over the 4 adjacent move rows (f16 in/out, f32 accumulate)
__global__ void sum4_kernel(const _Float16* __restrict__ in, _Float16* __restrict__ out) {
  long tid = (long)blockIdx.x * 256 + threadIdx.x;
  if (tid >= (long)PP * 256) return;
  long q = tid >> 8; int c = (int)(tid & 255);
  const _Float16* p = in + (size_t)q * 4 * 256 + c;
  out[tid] = (_Float16)((float)p[0] + (float)p[256] + (float)p[512] + (float)p[768]);
}

// Assemble pconcat [PP, 576] f16 (cols 567..575 zero padding)
__global__ void pconcat_kernel(const int* __restrict__ pInts, const float* __restrict__ pFeats,
                               const float* __restrict__ pE, const float* __restrict__ iE,
                               const float* __restrict__ aE, const float* __restrict__ tE,
                               const _Float16* __restrict__ moves, _Float16* __restrict__ out) {
  long tid = (long)blockIdx.x * 256 + threadIdx.x;
  if (tid >= (long)PP * 576) return;
  long r = tid / 576; int c = (int)(tid % 576);
  const int* q = pInts + r * 6;
  float v;
  if (c < 64)        v = pE[q[0] * 64 + c];
  else if (c < 128)  v = iE[q[1] * 64 + (c - 64)];
  else if (c < 256)  v = aE[q[2] * 128 + (c - 128)];
  else if (c < 286)  { int cc = c - 256; v = tE[q[3 + cc / 10] * 10 + cc % 10]; }
  else if (c < 311)  v = pFeats[r * 25 + (c - 286)];
  else if (c < 567)  v = (float)moves[r * 256 + (c - 311)];
  else               v = 0.f;
  out[tid] = (_Float16)v;
}

// logits[row] = x[row,:] . pool_W + pool_b   (one wave per row, f32 x)
__global__ void pool_logits_kernel(const float* __restrict__ x, const float* __restrict__ pw,
                                   const float* __restrict__ pb, float* __restrict__ lg) {
  int row = blockIdx.x * 8 + (threadIdx.x >> 5);
  int lane = threadIdx.x & 31;
  const float* xr = x + (size_t)row * 512;
  float s = 0.f;
  for (int i = lane; i < 512; i += 32) s += xr[i] * pw[i];
  s = wave_sum(s);
  if (lane == 0) lg[row] = s + pb[0];
}

// pooled[b,side,:] = sum_p softmax(logits)_p * pokes[b,side,p,:] -> xcat[:,0:1024]
__global__ void pooled_kernel(const float* __restrict__ x, const float* __restrict__ lg,
                              float* __restrict__ xcat) {
  int bs = blockIdx.x;   // b*2+side
  const float* l = lg + bs * 6;
  float m = -1e30f;
#pragma unroll
  for (int p = 0; p < 6; p++) m = fmaxf(m, l[p]);
  float e[6], s = 0.f;
#pragma unroll
  for (int p = 0; p < 6; p++) { e[p] = __expf(l[p] - m); s += e[p]; }
  float inv = 1.f / s;
  int b = bs >> 1, side = bs & 1;
  for (int d = threadIdx.x; d < 512; d += 256) {
    float acc = 0.f;
#pragma unroll
    for (int p = 0; p < 6; p++)
      acc += e[p] * inv * x[((size_t)bs * 6 + p) * 512 + d];
    xcat[(size_t)b * 1088 + side * 512 + d] = acc;
  }
}

// Batch-norm statistics: one block per feature, reduce over B rows (f32 in).
__global__ void bn_stats_kernel(const float* __restrict__ X, float* __restrict__ mean,
                                float* __restrict__ var, int Bn, int Dn) {
  int f = blockIdx.x;
  float s = 0.f, s2 = 0.f;
  for (int i = threadIdx.x; i < Bn; i += 256) {
    float v = X[(size_t)i * Dn + f];
    s += v; s2 += v * v;
  }
  __shared__ float sh[256], sh2[256];
  sh[threadIdx.x] = s; sh2[threadIdx.x] = s2;
  __syncthreads();
  for (int o = 128; o > 0; o >>= 1) {
    if (threadIdx.x < o) { sh[threadIdx.x] += sh[threadIdx.x + o]; sh2[threadIdx.x] += sh2[threadIdx.x + o]; }
    __syncthreads();
  }
  if (threadIdx.x == 0) {
    float mu = sh[0] / (float)Bn;
    mean[f] = mu;
    var[f] = sh2[0] / (float)Bn - mu * mu;
  }
}

__global__ void bn_apply_kernel(const float* __restrict__ X, const float* __restrict__ mean,
                                const float* __restrict__ var, const float* __restrict__ g,
                                const float* __restrict__ b, _Float16* __restrict__ Y,
                                long total, int Dn) {
  long tid = (long)blockIdx.x * 256 + threadIdx.x;
  if (tid >= total) return;
  int f = (int)(tid % Dn);
  Y[tid] = (_Float16)((X[tid] - mean[f]) * rsqrtf(var[f] + EPSF) * g[f] + b[f]);
}

__global__ void final_kernel(const float* __restrict__ x16, const float* __restrict__ fw,
                             const float* __restrict__ fb, float* __restrict__ out, int Bn) {
  int b = blockIdx.x * 256 + threadIdx.x;
  if (b >= Bn) return;
  float acc = fb[0];
#pragma unroll
  for (int i = 0; i < 16; i++) acc += x16[b * 16 + i] * fw[i];
  out[b] = acc;
}

// ---------------------------------------------------------------------------
// Host-side orchestration
// ---------------------------------------------------------------------------
extern "C" void kernel_launch(void* const* d_in, const int* in_sizes, int n_in,
                              void* d_out, int out_size, void* d_ws, size_t ws_size,
                              hipStream_t stream) {
  (void)in_sizes; (void)n_in; (void)out_size; (void)ws_size;

  // ---- inputs (setup_inputs dict order) ----
  const int*   boardInts  = (const int*)d_in[0];   // [B,5]
  const float* boardFeats = (const float*)d_in[1]; // [B,15]
  const int*   pokeInts   = (const int*)d_in[2];   // [B,2,6,6]
  const float* pokeFeats  = (const float*)d_in[3]; // [B,2,6,25]
  const int*   moveInts   = (const int*)d_in[4];   // [B,2,6,4,2]
  const float* moveFeats  = (const float*)d_in[5]; // [B,2,6,4,6]

  // ---- params, flattened in _init_params insertion order ----
  int pi = 6;
  auto NF = [&]() { return (const float*)d_in[pi++]; };
  const float* twEmb = NF();   const float* trEmb = NF();
  const float* weEmb = NF();   const float* teEmb = NF();
  const float* bLNg  = NF();   const float* bLNb  = NF();
  const float* boardW = NF();  const float* boardB = NF();
  const float* typeEmb = NF(); const float* moveEmb = NF();
  const float* Umat = NF();    const float* Wm = NF();
  const float* moveW = NF();   const float* moveB = NF();
  const float* pokeEmb = NF(); const float* itemEmb = NF();
  const float* abEmb = NF();
  const float* pokeW = NF();   const float* pokeB = NF();
  struct Blk {
    const float *ln1g, *ln1b, *Wq, *Wk, *Wv, *projW, *projb;
    const float *ln2g, *ln2b, *W1, *b1, *W2, *b2;
  } blk[3];
  for (int i = 0; i < 3; i++) {
    blk[i].ln1g = NF(); blk[i].ln1b = NF();
    blk[i].Wq = NF(); blk[i].Wk = NF(); blk[i].Wv = NF();
    blk[i].projW = NF(); blk[i].projb = NF();
    blk[i].ln2g = NF(); blk[i].ln2b = NF();
    blk[i].W1 = NF(); blk[i].b1 = NF();
    blk[i].W2 = NF(); blk[i].b2 = NF();
  }
  const float* poolW = NF();   const float* poolB = NF();
  struct Mlp { const float *g, *b, *W, *bias; } mlp[4];
  for (int i = 0; i < 4; i++) { mlp[i].g = NF(); mlp[i].b = NF(); mlp[i].W = NF(); mlp[i].bias = NF(); }
  const float* finalW = NF();  const float* finalB = NF();

  float* out = (float*)d_out;

  // ---- workspace bump allocator ----
  char* ws = (char*)d_ws;
  size_t off = 0;
  auto alloc = [&](size_t bytes) -> void* {
    size_t cur = (off + 255) & ~(size_t)255;
    off = cur + bytes;
    return (void*)(ws + cur);
  };
  auto fragHalves = [](int K, int N) { return (size_t)(K >> 5) * (N >> 4) * 512; };
  auto allocFrag = [&](int K, int N) { return (_Float16*)alloc(fragHalves(K, N) * 2); };
  auto allocH = [&](size_t n) { return (_Float16*)alloc(n * 2); };
  auto allocF = [&](size_t n) { return (float*)alloc(n * 4); };

  // packed f16 weight fragments
  _Float16* fMoveW = allocFrag(64, 256);
  _Float16* fPokeW = allocFrag(576, 512);
  _Float16 *fWq[3], *fWk[3], *fWv[3], *fProj[3], *fW1[3], *fW2[3];
  for (int i = 0; i < 3; i++) {
    fWq[i] = allocFrag(512, 512); fWk[i] = allocFrag(512, 512); fWv[i] = allocFrag(512, 512);
    fProj[i] = allocFrag(512, 512);
    fW1[i] = allocFrag(512, 1024); fW2[i] = allocFrag(1024, 512);
  }
  const int mlpDims[5] = {1088, 1024, 256, 64, 16};
  _Float16* fMlp[4];
  for (int i = 0; i < 4; i++) fMlp[i] = allocFrag(mlpDims[i], mlpDims[i + 1]);

  // activation buffers (f16 for GEMM inputs, f32 for residual stream & head)
  float*    typeU   = allocF((size_t)20 * 64);
  _Float16* mcomb   = allocH((size_t)RR * 64);
  _Float16* moveOut = allocH((size_t)RR * 256);
  _Float16* moves   = allocH((size_t)PP * 256);
  _Float16* pconcat = allocH((size_t)PP * 576);
  float*    xA      = allocF((size_t)PP * 512);
  float*    xBf     = allocF((size_t)PP * 512);
  _Float16* xn      = allocH((size_t)PP * 512);
  _Float16* qb      = allocH((size_t)PP * 512);
  _Float16* kb      = allocH((size_t)PP * 512);
  _Float16* vb      = allocH((size_t)PP * 512);
  _Float16* attnO   = allocH((size_t)PP * 512);
  _Float16* hbuf    = allocH((size_t)PP * 1024);
  float*    logits  = allocF((size_t)PP);
  float*    xcat    = allocF((size_t)BB * 1088);
  _Float16* bnbuf   = allocH((size_t)BB * 1088);
  float*    bmean   = allocF(1088);
  float*    bvar    = allocF(1088);
  float* ybuf[4];
  for (int i = 0; i < 4; i++) ybuf[i] = allocF((size_t)BB * mlpDims[i + 1]);

  // ---- launch helpers ----
  auto packW = [&](const float* W, _Float16* F, int K, int N, int Kreal) {
    int total = (K >> 5) * (N >> 4) * 32;
    pack_weight_kernel<<<(total + 255) / 256, 256, 0, stream>>>(W, F, K, N, Kreal);
  };
  auto gemm = [&](const _Float16* A, const _Float16* F, const float* bias,
                  const float* resid, float* C32, _Float16* C16,
                  int M, int N, int K, int relu) {
    int nTiles = N >> 4;
    if ((nTiles & 3) == 0) {
      long tiles = (long)(M >> 4) * (nTiles >> 2);
      gemm_f16_kernel<4><<<(unsigned)((tiles + 7) >> 3), 256, 0, stream>>>(
          A, F, bias, resid, C32, C16, M, N, K, relu);
    } else {
      long tiles = (long)(M >> 4) * nTiles;
      gemm_f16_kernel<1><<<(unsigned)((tiles + 7) >> 3), 256, 0, stream>>>(
          A, F, bias, resid, C32, C16, M, N, K, relu);
    }
  };

  // ---- 0. pack all weights to f16 fragments ----
  packW(moveW, fMoveW, 64, 256, 64);
  packW(pokeW, fPokeW, 576, 512, 567);
  for (int i = 0; i < 3; i++) {
    pack_qkv_kernel<<<64, 256, 0, stream>>>(blk[i].Wq, fWq[i]);
    pack_qkv_kernel<<<64, 256, 0, stream>>>(blk[i].Wk, fWk[i]);
    pack_qkv_kernel<<<64, 256, 0, stream>>>(blk[i].Wv, fWv[i]);
    packW(blk[i].projW, fProj[i], 512, 512, 512);
    packW(blk[i].W1, fW1[i], 512, 1024, 512);
    packW(blk[i].W2, fW2[i], 1024, 512, 1024);
  }
  for (int i = 0; i < 4; i++) packW(mlp[i].W, fMlp[i], mlpDims[i], mlpDims[i + 1], mlpDims[i]);

  // ---- 1. board encoder -> xcat[:, 1024:1088] ----
  board_kernel<<<BB, 64, 0, stream>>>(boardInts, boardFeats, twEmb, trEmb, weEmb, teEmb,
                                      bLNg, bLNb, boardW, boardB, xcat);

  // ---- 2. move encoder ----
  type_u_kernel<<<5, 256, 0, stream>>>(typeEmb, Umat, typeU);
  {
    long tot = (long)RR * 64;
    mcomb_kernel<<<(unsigned)((tot + 255) / 256), 256, 0, stream>>>(
        moveInts, moveFeats, moveEmb, typeU, Wm, mcomb);
  }
  gemm(mcomb, fMoveW, moveB, nullptr, nullptr, moveOut, RR, 256, 64, 1);
  {
    long tot = (long)PP * 256;
    sum4_kernel<<<(unsigned)((tot + 255) / 256), 256, 0, stream>>>(moveOut, moves);
  }

  // ---- 3. poke encoder -> x [PP, 512] f32 residual stream ----
  {
    long tot = (long)PP * 576;
    pconcat_kernel<<<(unsigned)((tot + 255) / 256), 256, 0, stream>>>(
        pokeInts, pokeFeats, pokeEmb, itemEmb, abEmb, typeEmb, moves, pconcat);
  }
  gemm(pconcat, fPokeW, pokeB, nullptr, xA, nullptr, PP, 512, 576, 1);

  // ---- 4. transformer blocks ----
  float* xCur = xA;
  float* xAlt = xBf;
  for (int i = 0; i < 3; i++) {
    layernorm_kernel<<<PP / 8, 256, 0, stream>>>(xCur, blk[i].ln1g, blk[i].ln1b, xn, 512);
    gemm(xn, fWq[i], nullptr, nullptr, nullptr, qb, PP, 512, 512, 0);
    gemm(xn, fWk[i], nullptr, nullptr, nullptr, kb, PP, 512, 512, 0);
    gemm(xn, fWv[i], nullptr, nullptr, nullptr, vb, PP, 512, 512, 0);
    attention_kernel<<<BB * 8, 256, 0, stream>>>(qb, kb, vb, attnO);
    gemm(attnO, fProj[i], blk[i].projb, xCur, xAlt, nullptr, PP, 512, 512, 0);
    { float* t = xCur; xCur = xAlt; xAlt = t; }
    layernorm_kernel<<<PP / 8, 256, 0, stream>>>(xCur, blk[i].ln2g, blk[i].ln2b, xn, 512);
    gemm(xn, fW1[i], blk[i].b1, nullptr, nullptr, hbuf, PP, 1024, 512, 1);
    gemm(hbuf, fW2[i], blk[i].b2, xCur, xAlt, nullptr, PP, 512, 1024, 0);
    { float* t = xCur; xCur = xAlt; xAlt = t; }
  }

  // ---- 5. attention pooling -> xcat[:, 0:1024] ----
  pool_logits_kernel<<<PP / 8, 256, 0, stream>>>(xCur, poolW, poolB, logits);
  pooled_kernel<<<BB * 2, 256, 0, stream>>>(xCur, logits, xcat);

  // ---- 6. BN-MLP head ----
  const float* mlpIn = xcat;
  for (int i = 0; i < 4; i++) {
    int Dn = mlpDims[i];
    long tot = (long)BB * Dn;
    bn_stats_kernel<<<Dn, 256, 0, stream>>>(mlpIn, bmean, bvar, BB, Dn);
    bn_apply_kernel<<<(unsigned)((tot + 255) / 256), 256, 0, stream>>>(
        mlpIn, bmean, bvar, mlp[i].g, mlp[i].b, bnbuf, tot, Dn);
    gemm(bnbuf, fMlp[i], mlp[i].bias, nullptr, ybuf[i], nullptr, BB, mlpDims[i + 1], Dn, 1);
    mlpIn = ybuf[i];
  }

  // ---- 7. final 16 -> 1 ----
  final_kernel<<<(BB + 255) / 256, 256, 0, stream>>>(ybuf[3], finalW, finalB, out, BB);
}
